// TwoGNN_37177236914920
// MI455X (gfx1250) — matrix operand order repaired and checked by reading, not compiled, not run on hardware
//
#include <hip/hip_runtime.h>
#include <hip/hip_bf16.h>

// ---------------------------------------------------------------------------
// CDNA5 (gfx1250) GNN forward:
//   3x [ self = h@W1 ; y = h@W2 ; acc[dst] += y[src] ; h = relu(self+acc) ]
//   then segment-sum pool + 2-layer MLP head.
// GEMMs: bf16 inputs, f32 accumulate via V_WMMA_F32_16X16X32_BF16.
// Block tile 128x128 (8 waves, wave tile 32x64 = 8 WMMA / K-step).
// Global->LDS staging uses gfx1250 async-to-LDS double buffering
// (GLOBAL_LOAD_ASYNC_TO_LDS_B128 + s_wait_asynccnt), with a reg-staged
// fallback if the builtins are absent.
// ---------------------------------------------------------------------------

typedef __attribute__((ext_vector_type(16))) __bf16 v16bf;
typedef __attribute__((ext_vector_type(8)))  __bf16 v8bf;
typedef __attribute__((ext_vector_type(4)))  __bf16 v4bf;
typedef __attribute__((ext_vector_type(8)))  float  v8f;
typedef __attribute__((ext_vector_type(4)))  int    v4i;

#define NSETS   100000
#define NEDGES  400000
#define IN_DIM  129
#define K0PAD   160     // IN_DIM padded to multiple of 32
#define HID     512
#define ODIM    10
#define NGRAPH  64

#if defined(__has_builtin)
#  if __has_builtin(__builtin_amdgcn_global_load_async_to_lds_b128) && \
      __has_builtin(__builtin_amdgcn_s_wait_asynccnt)
#    define HAVE_ASYNC 1
#  endif
#endif
#ifndef HAVE_ASYNC
#  define HAVE_ASYNC 0
#endif

static __device__ __forceinline__ __bf16 f2bf(float f) {
    unsigned u = __builtin_bit_cast(unsigned, f);
    u += 0x7FFFu + ((u >> 16) & 1u);          // round-to-nearest-even
    unsigned short h = (unsigned short)(u >> 16);
    return __builtin_bit_cast(__bf16, h);
}

#if HAVE_ASYNC
static __device__ __forceinline__ void cp_async16(const void* g, void* l) {
    __builtin_amdgcn_global_load_async_to_lds_b128(
        (__attribute__((address_space(1))) v4i*)(unsigned long long)g,
        (__attribute__((address_space(3))) v4i*)(unsigned)(unsigned long long)l,
        0, 0);
}
#endif

// ---- x2 [S,129] f32 -> [S,160] bf16 (zero-padded K) -----------------------
__global__ void k_cvt_x2(const float* __restrict__ x2, __bf16* __restrict__ hb0) {
    int i = blockIdx.x * 256 + threadIdx.x;
    if (i >= NSETS * K0PAD) return;
    int m = i / K0PAD, k = i - m * K0PAD;
    hb0[i] = (k < IN_DIM) ? f2bf(x2[m * IN_DIM + k]) : f2bf(0.0f);
}

// ---- W [K,N] f32 -> Wt [N,Kpad] bf16 (transpose + pad) --------------------
__global__ void k_wt(const float* __restrict__ W, __bf16* __restrict__ Wt,
                     int K, int N, int Kpad) {
    int i = blockIdx.x * 256 + threadIdx.x;
    if (i >= N * Kpad) return;
    int n = i / Kpad, k = i - n * Kpad;
    Wt[i] = (k < K) ? f2bf(W[k * N + n]) : f2bf(0.0f);
}

// ---- WMMA GEMM: C[M,512] f32 = A[M,Kpad] bf16 * Wt[512,Kpad]^T ------------
__global__ void __launch_bounds__(256)
k_gemm_wmma(const __bf16* __restrict__ A, const __bf16* __restrict__ Bt,
            float* __restrict__ C, int M, int Kpad) {
#if HAVE_ASYNC
    __shared__ __bf16 sA[2][128 * 32];   // 2 x 8 KB
    __shared__ __bf16 sB[2][128 * 32];   // 2 x 8 KB
#else
    __shared__ __bf16 sA1[128 * 32];
    __shared__ __bf16 sB1[128 * 32];
#endif
    const int tid  = threadIdx.x;
    const int lane = tid & 31;
    const int wave = tid >> 5;
    const int wm   = (wave & 3) * 32;     // 4 waves along M (128)
    const int wn   = (wave >> 2) * 64;    // 2 waves along N (128)
    const int mblk = blockIdx.x * 128;
    const int nblk = blockIdx.y * 128;
    const int lm   = lane & 15;
    const int lhi  = lane >> 4;
    // staging map: thread copies chunks (r0,c0) and (r0+64,c0) of each tile
    const int r0 = tid >> 2;
    const int c0 = (tid & 3) * 8;

    v8f acc[2][4] = {};

    // one 32-K-step tile of WMMA work from LDS buffers pA/pB
    auto do_tile = [&](const __bf16* pA, const __bf16* pB) {
        v16bf a[2], b[4];
        const int kb  = lhi * 8;
        const int kb2 = lhi * 16;
#pragma unroll
        for (int mi = 0; mi < 2; ++mi) {
            const __bf16* p = pA + (wm + mi * 16 + lm) * 32;
            a[mi] = *(const v16bf*)(p + kb2);
        }
#pragma unroll
        for (int ni = 0; ni < 4; ++ni)
            b[ni] = *(const v16bf*)(pB + (wn + ni * 16 + lm) * 32 + kb2);
#pragma unroll
        for (int mi = 0; mi < 2; ++mi)
#pragma unroll
            for (int ni = 0; ni < 4; ++ni)
                acc[mi][ni] = __builtin_amdgcn_wmma_f32_16x16x32_bf16(
                    false, a[mi], false, b[ni], (short)0, acc[mi][ni], false, false);
    };

#if HAVE_ASYNC
    // Pre-zero A rows beyond M once: those lanes never issue async copies,
    // so the zeros persist for every K-step and EXEC stays uniform at WMMAs.
    {
        v8bf z = {};
        if (mblk + r0 >= M) {
            *(v8bf*)&sA[0][r0 * 32 + c0] = z;
            *(v8bf*)&sA[1][r0 * 32 + c0] = z;
        }
        if (mblk + r0 + 64 >= M) {
            *(v8bf*)&sA[0][(r0 + 64) * 32 + c0] = z;
            *(v8bf*)&sA[1][(r0 + 64) * 32 + c0] = z;
        }
    }
    auto stage = [&](int buf, int k0) {
        if (mblk + r0 < M)
            cp_async16(A + (long)(mblk + r0) * Kpad + k0 + c0,
                       &sA[buf][r0 * 32 + c0]);
        if (mblk + r0 + 64 < M)
            cp_async16(A + (long)(mblk + r0 + 64) * Kpad + k0 + c0,
                       &sA[buf][(r0 + 64) * 32 + c0]);
        cp_async16(Bt + (long)(nblk + r0) * Kpad + k0 + c0,
                   &sB[buf][r0 * 32 + c0]);
        cp_async16(Bt + (long)(nblk + r0 + 64) * Kpad + k0 + c0,
                   &sB[buf][(r0 + 64) * 32 + c0]);
    };
    stage(0, 0);
    __builtin_amdgcn_s_wait_asynccnt(0);
    __syncthreads();
    int buf = 0;
    for (int k0 = 0; k0 < Kpad; k0 += 32) {
        if (k0 + 32 < Kpad) stage(buf ^ 1, k0 + 32);   // prefetch next tile
        do_tile(&sA[buf][0], &sB[buf][0]);             // compute current tile
        __builtin_amdgcn_s_wait_asynccnt(0);           // next tile landed
        __syncthreads();                               // all readers done
        buf ^= 1;
    }
#else
    for (int k0 = 0; k0 < Kpad; k0 += 32) {
        v8bf va0 = {}, va1 = {};
        if (mblk + r0 < M)
            va0 = *(const v8bf*)(A + (long)(mblk + r0) * Kpad + k0 + c0);
        if (mblk + r0 + 64 < M)
            va1 = *(const v8bf*)(A + (long)(mblk + r0 + 64) * Kpad + k0 + c0);
        v8bf vb0 = *(const v8bf*)(Bt + (long)(nblk + r0) * Kpad + k0 + c0);
        v8bf vb1 = *(const v8bf*)(Bt + (long)(nblk + r0 + 64) * Kpad + k0 + c0);
        __syncthreads();
        *(v8bf*)&sA1[r0 * 32 + c0] = va0;
        *(v8bf*)&sA1[(r0 + 64) * 32 + c0] = va1;
        *(v8bf*)&sB1[r0 * 32 + c0] = vb0;
        *(v8bf*)&sB1[(r0 + 64) * 32 + c0] = vb1;
        __syncthreads();
        do_tile(sA1, sB1);
    }
#endif

    // ---- store: C/D layout: vgpr r -> (M = base+r, N = lane&15), lane>=16 -> M+8
    {
        const int col = nblk + wn + lm;
        const int rb  = mblk + wm + lhi * 8;
#pragma unroll
        for (int mi = 0; mi < 2; ++mi)
#pragma unroll
            for (int r = 0; r < 8; ++r) {
                int row = rb + mi * 16 + r;
                if (row < M) {
                    float* cp = C + (long)row * HID + col;
                    cp[0]  = acc[mi][0][r];
                    cp[16] = acc[mi][1][r];
                    cp[32] = acc[mi][2][r];
                    cp[48] = acc[mi][3][r];
                }
            }
    }
}

// ---- edge scatter: acc[dst] += y[src], float4 per thread ------------------
__global__ void __launch_bounds__(128)
k_scatter(const long long* __restrict__ ei, const float* __restrict__ y,
          float* __restrict__ acc) {
    int e = blockIdx.x;
    int t = threadIdx.x;                          // 128 threads x float4 = 512
    long long s = ei[e];
    long long d = ei[NEDGES + e];
    float4 v = ((const float4*)(y + s * HID))[t];
    float* b = acc + d * HID + t * 4;
    atomicAdd(b + 0, v.x);
    atomicAdd(b + 1, v.y);
    atomicAdd(b + 2, v.z);
    atomicAdd(b + 3, v.w);
}

// ---- relu in place + bf16 copy for next layer (float4 / v4bf per thread) --
__global__ void k_relu_bf16(float* __restrict__ acc, __bf16* __restrict__ hb, int n4) {
    int i = blockIdx.x * 256 + threadIdx.x;
    if (i >= n4) return;
    float4 v = ((float4*)acc)[i];
    v.x = fmaxf(v.x, 0.0f);
    v.y = fmaxf(v.y, 0.0f);
    v.z = fmaxf(v.z, 0.0f);
    v.w = fmaxf(v.w, 0.0f);
    ((float4*)acc)[i] = v;
    v4bf o;
    o[0] = f2bf(v.x); o[1] = f2bf(v.y); o[2] = f2bf(v.z); o[3] = f2bf(v.w);
    ((v4bf*)hb)[i] = o;
}

__global__ void k_zero(float* __restrict__ p, int n) {
    int i = blockIdx.x * 256 + threadIdx.x;
    if (i < n) p[i] = 0.0f;
}

// ---- segment-sum pool over sorted batch ids -------------------------------
__global__ void k_pool(const float* __restrict__ h, const long long* __restrict__ batch,
                       float* __restrict__ g, int n) {
    int i = blockIdx.x * 256 + threadIdx.x;
    if (i >= n) return;
    int m = i >> 9;            // /512
    int j = i & 511;
    atomicAdd(&g[(int)batch[m] * HID + j], h[i]);
}

// ---- classifier head (tiny: 64x512x512 + 64x512x10) -----------------------
__global__ void k_cls1(const float* __restrict__ g, const float* __restrict__ w1,
                       const float* __restrict__ b1, float* __restrict__ g2) {
    int i = blockIdx.x * 256 + threadIdx.x;
    if (i >= NGRAPH * HID) return;
    int m = i >> 9, n = i & 511;
    float s = b1[n];
    const float* gr = g + m * HID;
    for (int k = 0; k < HID; ++k) s += gr[k] * w1[k * HID + n];
    g2[i] = s > 0.0f ? s : 0.0f;
}

__global__ void k_cls2(const float* __restrict__ g2, const float* __restrict__ w2,
                       const float* __restrict__ b2, float* __restrict__ out) {
    int i = blockIdx.x * 64 + threadIdx.x;
    if (i >= NGRAPH * ODIM) return;
    int m = i / ODIM, o = i - m * ODIM;
    float s = b2[o];
    const float* gr = g2 + m * HID;
    for (int k = 0; k < HID; ++k) s += gr[k] * w2[k * ODIM + o];
    out[i] = s;
}

// ---------------------------------------------------------------------------
extern "C" void kernel_launch(void* const* d_in, const int* in_sizes, int n_in,
                              void* d_out, int out_size, void* d_ws, size_t ws_size,
                              hipStream_t stream) {
    const float*     x2    = (const float*)d_in[0];
    const long long* ei    = (const long long*)d_in[1];
    const long long* batch = (const long long*)d_in[2];
    const float* W1[3] = { (const float*)d_in[3], (const float*)d_in[5], (const float*)d_in[7] };
    const float* W2[3] = { (const float*)d_in[4], (const float*)d_in[6], (const float*)d_in[8] };
    const float* cw1 = (const float*)d_in[9];
    const float* cb1 = (const float*)d_in[10];
    const float* cw2 = (const float*)d_in[11];
    const float* cb2 = (const float*)d_in[12];
    float* out = (float*)d_out;

    // workspace carve-out (256B aligned)
    char*  ws  = (char*)d_ws;
    size_t off = 0;
    auto alloc = [&](size_t bytes) -> void* {
        void* p = ws + off;
        off = (off + bytes + 255) & ~(size_t)255;
        return p;
    };
    __bf16* hb0  = (__bf16*)alloc((size_t)NSETS * K0PAD * 2);   //  32 MB
    __bf16* hb   = (__bf16*)alloc((size_t)NSETS * HID * 2);     // 102 MB (fits L2)
    float*  accb = (float*)alloc((size_t)NSETS * HID * 4);      // 205 MB
    float*  ybuf = (float*)alloc((size_t)NSETS * HID * 4);      // 205 MB
    __bf16* Wt1[3], * Wt2[3];
    Wt1[0] = (__bf16*)alloc((size_t)HID * K0PAD * 2);
    Wt2[0] = (__bf16*)alloc((size_t)HID * K0PAD * 2);
    for (int l = 1; l < 3; ++l) {
        Wt1[l] = (__bf16*)alloc((size_t)HID * HID * 2);
        Wt2[l] = (__bf16*)alloc((size_t)HID * HID * 2);
    }
    float* g  = (float*)alloc((size_t)NGRAPH * HID * 4);
    float* g2 = (float*)alloc((size_t)NGRAPH * HID * 4);

    // ---- prep: convert activations & weights to bf16 (transposed) ----
    k_cvt_x2<<<(NSETS * K0PAD + 255) / 256, 256, 0, stream>>>(x2, hb0);
    k_wt<<<(HID * K0PAD + 255) / 256, 256, 0, stream>>>(W1[0], Wt1[0], IN_DIM, HID, K0PAD);
    k_wt<<<(HID * K0PAD + 255) / 256, 256, 0, stream>>>(W2[0], Wt2[0], IN_DIM, HID, K0PAD);
    for (int l = 1; l < 3; ++l) {
        k_wt<<<(HID * HID + 255) / 256, 256, 0, stream>>>(W1[l], Wt1[l], HID, HID, HID);
        k_wt<<<(HID * HID + 255) / 256, 256, 0, stream>>>(W2[l], Wt2[l], HID, HID, HID);
    }

    const dim3 gg((NSETS + 127) / 128, HID / 128);
    const int  nElem = NSETS * HID;

    // ---- layer 0 (K=160) ----
    k_gemm_wmma<<<gg, 256, 0, stream>>>(hb0, Wt1[0], accb, NSETS, K0PAD);
    k_gemm_wmma<<<gg, 256, 0, stream>>>(hb0, Wt2[0], ybuf, NSETS, K0PAD);
    k_scatter<<<NEDGES, 128, 0, stream>>>(ei, ybuf, accb);
    k_relu_bf16<<<(nElem / 4 + 255) / 256, 256, 0, stream>>>(accb, hb, nElem / 4);

    // ---- layers 1..2 (K=512) ----
    for (int l = 1; l < 3; ++l) {
        k_gemm_wmma<<<gg, 256, 0, stream>>>(hb, Wt1[l], accb, NSETS, HID);
        k_gemm_wmma<<<gg, 256, 0, stream>>>(hb, Wt2[l], ybuf, NSETS, HID);
        k_scatter<<<NEDGES, 128, 0, stream>>>(ei, ybuf, accb);
        k_relu_bf16<<<(nElem / 4 + 255) / 256, 256, 0, stream>>>(accb, hb, nElem / 4);
    }

    // ---- pool + head ----
    k_zero<<<(NGRAPH * HID + 255) / 256, 256, 0, stream>>>(g, NGRAPH * HID);
    k_pool<<<(nElem + 255) / 256, 256, 0, stream>>>(accb, batch, g, nElem);
    k_cls1<<<(NGRAPH * HID + 255) / 256, 256, 0, stream>>>(g, cw1, cb1, g2);
    k_cls2<<<(NGRAPH * ODIM + 63) / 64, 64, 0, stream>>>(g2, cw2, cb2, out);
}
